// MultiHeadAttention_patch2_6227702579733
// MI455X (gfx1250) — matrix operand order repaired and checked by reading, not compiled, not run on hardware
//
#include <hip/hip_runtime.h>

// ---------------------------------------------------------------------------
// MI455X (gfx1250) fused patch-attention.
//
//   Z = XR @ [G | Mvf]   via v_wmma_f32_16x16x32_f16  (the only large GEMM)
//   G   = SCALE * Wq^T Wk          (55x55)  -> attention logits L = xr G xr^T
//   Mvf = Wv^T Wfc^T               (55x55)  -> Y = xr Mvf ; out = softmax(L) Y + b
//
// Memory-bound target: 461 MB @ 23.3 TB/s ~= 20 us; WMMA makes compute free.
// P2 (per-group attention tail) is split 2-threads-per-row so all 4 waves of
// the workgroup share it, keeping it under the HBM budget.
// ---------------------------------------------------------------------------

typedef __attribute__((ext_vector_type(16))) _Float16 v16h;
typedef __attribute__((ext_vector_type(8)))  float    v8f;

#define NGROUPS  (4096 * 64)   // B*C = 262144 independent attention groups
#define GPW      16            // groups per workgroup
#define THREADS  128           // 4 waves (wave32)
#define ZSTRIDE  116           // LDS row stride for Z tile (bank-conflict free)

// ---------------------------------------------------------------------------
// Setup kernel: build [G | Mvf] (55x110, padded to 64x112) as f16, stored
// directly in WMMA B-fragment order:
//   frag(kk,nt) : 32 lanes x 16 halves, lane L holds col N = nt*16 + (L&15),
//   half j holds K = kk*32 + (L<16 ? 0 : 16) + j   (contiguous k order, same as A).
// Total 14 frags * 1 KB = 14,336 bytes in d_ws.
// ---------------------------------------------------------------------------
__global__ __launch_bounds__(256) void setup_wfrags(
    const float* __restrict__ wq, const float* __restrict__ wk,
    const float* __restrict__ wv, const float* __restrict__ wfc,
    _Float16* __restrict__ wfrag)
{
  const float SCALE = 0.14433756729740643f;  // 1/sqrt(48)
  int i = blockIdx.x * blockDim.x + threadIdx.x;
  if (i >= 2 * 7 * 32 * 16) return;
  int j2   = i & 15;          // half index within lane's fragment
  int L    = (i >> 4) & 31;   // lane
  int frag = i >> 9;          // 0..13
  int kk = frag / 7, nt = frag % 7;
  int N = nt * 16 + (L & 15);                    // 0..111 (col of [G|Mvf])
  int K = kk * 32 + ((L < 16) ? 0 : 16) + j2;    // 0..63  (row, = d index)
  float val = 0.0f;
  if (K < 55 && N < 110) {
    float acc = 0.0f;
    if (N < 55) {        // G[K][N] = SCALE * sum_e wq[e,K] * wk[e,N]
      for (int e = 0; e < 48; ++e) acc += wq[e * 55 + K] * wk[e * 55 + N];
      val = acc * SCALE;
    } else {             // Mvf[K][N-55] = sum_e wv[e,K] * wfc[N-55, e]
      int dout = N - 55;
      for (int e = 0; e < 48; ++e) acc += wv[e * 55 + K] * wfc[dout * 48 + e];
      val = acc;
    }
  }
  wfrag[i] = (_Float16)val;
}

// ---------------------------------------------------------------------------
// Main fused kernel. One workgroup = 4 waves = 64 rows = 16 groups.
// ---------------------------------------------------------------------------
__global__ __launch_bounds__(THREADS) void mha_fused(
    const float* __restrict__ x, const float* __restrict__ bfc,
    const _Float16* __restrict__ wfrag, float* __restrict__ out)
{
  __shared__ float xlds[GPW * 220];          // staged input images (f32)
  __shared__ float zlds[64 * ZSTRIDE];       // Z = [t | Y] tiles, 64 rows x 112
  __shared__ float outlds[GPW * 220];        // output images (image layout)
  __shared__ float llds[64 * 4];             // attention logits L[row][4]
  __shared__ float bflds[64];

  const int tid = threadIdx.x;
  const size_t gbase = (size_t)blockIdx.x * (GPW * 220);

  // ---- P0: coalesced global -> LDS (16 contiguous 220-float group images) --
  {
    const float4* xg = (const float4*)(x + gbase);
    float4* xl = (float4*)xlds;
    for (int i = tid; i < GPW * 220 / 4; i += THREADS) xl[i] = xg[i];
    if (tid < 64) bflds[tid] = (tid < 55) ? bfc[tid] : 0.0f;
  }
  __syncthreads();

  // ---- P1: WMMA  Z(64x112) = A(64x64,f16) x W(64x112,f16) ------------------
  {
    const int wave = tid >> 5, lane = tid & 31;
    const int laneHi = lane >> 4;     // A layout: lanes 16-31 hold K+16..K+31
    const int M = lane & 15;          // row within 16-row tile
    const int R = wave * 16 + M;      // local row: group gi, patch s
    const int gi = R >> 2, s = R & 3;
    // patch (p1,p2) origin inside the 22x10 image
    const int pbase = gi * 220 + (s >> 1) * 110 + (s & 1) * 5;

    // Gather A fragments (k-step 0: d=0..31, k-step 1: d=32..63 zero-padded),
    // converting f32 -> f16. Contiguous k order (k = 16*laneHi + j), as in B.
    v16h a0, a1;
#pragma unroll
    for (int j = 0; j < 16; ++j) {
      const int v = j >> 1, w = j & 1;
      const int Kb = laneHi * 16 + j; // 0..31
      a0[j] = (_Float16)xlds[pbase + (Kb / 5) * 10 + (Kb % 5)];
      const int K1 = 32 + Kb;                                              // 32..63
      a1[j] = (K1 < 55) ? (_Float16)xlds[pbase + (K1 / 5) * 10 + (K1 % 5)]
                        : (_Float16)0.0f;
    }

    const v16h* wf = (const v16h*)wfrag;   // 32 B/lane, coalesced, L2-resident
    const int col  = lane & 15;
    const int rowb = wave * 16 + laneHi * 8;
#pragma unroll
    for (int nt = 0; nt < 7; ++nt) {
      v8f c = {};
      v16h b0 = wf[nt * 32 + lane];
      c = __builtin_amdgcn_wmma_f32_16x16x32_f16(false, a0, false, b0,
                                                 (short)0, c, false, false);
      v16h b1 = wf[(7 + nt) * 32 + lane];
      c = __builtin_amdgcn_wmma_f32_16x16x32_f16(false, a1, false, b1,
                                                 (short)0, c, false, false);
      // D layout: VGPR i -> row (rowb+i), col nt*16 + lane%16
#pragma unroll
      for (int i2 = 0; i2 < 8; ++i2)
        zlds[(rowb + i2) * ZSTRIDE + nt * 16 + col] = c[i2];
    }
  }
  __syncthreads();

  // ---- P2a: logits, 2 threads per row; each computes 2 of 4 dots -----------
  {
    const int r = tid >> 1;            // row 0..63  -> (gi, s)
    const int half = tid & 1;
    const int gi = r >> 2;
    const float* trow = &zlds[r * ZSTRIDE];     // t[row, 0..54]
#pragma unroll
    for (int u = 0; u < 2; ++u) {
      const int t2 = 2 * half + u;     // which key-patch
      const float* xp = &xlds[gi * 220 + (t2 >> 1) * 110 + (t2 & 1) * 5];
      float acc = 0.0f;
      int d = 0;
      for (int h = 0; h < 11; ++h)
        for (int w = 0; w < 5; ++w, ++d) acc += trow[d] * xp[h * 10 + w];
      llds[r * 4 + t2] = acc;
    }
  }
  __syncthreads();

  // ---- P2b: softmax (redundant per pair) + half of the output row ----------
  {
    const int r = tid >> 1;
    const int half = tid & 1;
    const int gi = r >> 2, s = r & 3;

    const float4 lv = *(const float4*)&llds[r * 4];
    const float m = fmaxf(fmaxf(lv.x, lv.y), fmaxf(lv.z, lv.w));
    float p0 = __expf(lv.x - m), p1 = __expf(lv.y - m);
    float p2 = __expf(lv.z - m), p3 = __expf(lv.w - m);
    const float inv = 1.0f / (p0 + p1 + p2 + p3);
    p0 *= inv; p1 *= inv; p2 *= inv; p3 *= inv;

    const float* y0 = &zlds[(gi * 4 + 0) * ZSTRIDE + 55];
    const float* y1 = &zlds[(gi * 4 + 1) * ZSTRIDE + 55];
    const float* y2 = &zlds[(gi * 4 + 2) * ZSTRIDE + 55];
    const float* y3 = &zlds[(gi * 4 + 3) * ZSTRIDE + 55];
    float* op = &outlds[gi * 220 + (s >> 1) * 110 + (s & 1) * 5];
    const int h0 = half ? 6 : 0;
    const int h1 = half ? 11 : 6;
    for (int h = h0; h < h1; ++h)
      for (int w = 0; w < 5; ++w) {
        const int d = h * 5 + w;       // element index within patch
        op[h * 10 + w] = p0 * y0[d] + p1 * y1[d] + p2 * y2[d] + p3 * y3[d]
                         + bflds[d];
      }
  }
  __syncthreads();

  // ---- P3: coalesced LDS -> global store (output layout == input layout) ---
  {
    float4* og = (float4*)(out + gbase);
    const float4* ol = (const float4*)outlds;
    for (int i = tid; i < GPW * 220 / 4; i += THREADS) og[i] = ol[i];
  }
}

// ---------------------------------------------------------------------------
extern "C" void kernel_launch(void* const* d_in, const int* in_sizes, int n_in,
                              void* d_out, int out_size, void* d_ws, size_t ws_size,
                              hipStream_t stream) {
  (void)in_sizes; (void)n_in; (void)out_size; (void)ws_size;
  const float* x   = (const float*)d_in[0];
  const float* wq  = (const float*)d_in[1];
  const float* wk  = (const float*)d_in[2];
  const float* wv  = (const float*)d_in[3];
  const float* wfc = (const float*)d_in[4];
  const float* bfc = (const float*)d_in[5];
  float*    out    = (float*)d_out;
  _Float16* wfrag  = (_Float16*)d_ws;   // 14,336 bytes used

  hipLaunchKernelGGL(setup_wfrags, dim3(28), dim3(256), 0, stream,
                     wq, wk, wv, wfc, wfrag);
  hipLaunchKernelGGL(mha_fused, dim3(NGROUPS / GPW), dim3(THREADS), 0, stream,
                     x, bfc, wfrag, out);
}
